// GraphSAGE_15985868276246
// MI455X (gfx1250) — matrix operand order repaired and checked by reading, not compiled, not run on hardware
//
#include <hip/hip_runtime.h>
#include <hip/hip_bf16.h>

typedef __bf16 bf16_t;
typedef __attribute__((ext_vector_type(8)))  __bf16 v8bf;
typedef __attribute__((ext_vector_type(16))) __bf16 v16bf;
typedef __attribute__((ext_vector_type(8)))  float  v8f;
typedef __attribute__((ext_vector_type(4)))  int    v4i;

#define GLOBAL_AS __attribute__((address_space(1)))
#define LDS_AS    __attribute__((address_space(3)))

#if __has_builtin(__builtin_amdgcn_global_load_async_to_lds_b128)
#define HAVE_ASYNC_LDS 1
#else
#define HAVE_ASYNC_LDS 0
#endif

// LDS row strides (in bf16 elements), padded to avoid bank conflicts:
// 80B / 144B row pitches are 16B-aligned but not multiples of the 256B
// bank period, unlike the natural 64B/128B pitches.
#define LDAK 40
#define LDBN 72

// Copy 16 bytes global -> LDS. Async (ASYNCcnt) when the builtin exists.
__device__ __forceinline__ void async_copy16(const void* g, void* l) {
#if HAVE_ASYNC_LDS
  __builtin_amdgcn_global_load_async_to_lds_b128(
      (GLOBAL_AS v4i*)g, (LDS_AS v4i*)l, 0, 0);
#else
  *(uint4*)l = *(const uint4*)g;
#endif
}

__device__ __forceinline__ void async_wait0() {
#if __has_builtin(__builtin_amdgcn_s_wait_asynccnt)
  __builtin_amdgcn_s_wait_asynccnt(0);
#elif HAVE_ASYNC_LDS
  asm volatile("s_wait_asynccnt 0" ::: "memory");
#endif
}

// ---- WMMA fragment loaders (CDNA5 16-bit layouts, ISA 7.12.2) -------------
// A: 16x32 (MxK) bf16 tile, LDS row-major, leading dim ldk (halves).
__device__ __forceinline__ v16bf load_frag_a(const bf16_t* lds, int row0, int ldk) {
  const int lane  = threadIdx.x & 31;
  const int m     = lane & 15;
  const int kbase = (lane < 16) ? 0 : 8;
  const bf16_t* p = lds + (row0 + m) * ldk;
  v8bf lo = *(const v8bf*)(p + kbase);        // K = kbase .. kbase+7
  v8bf hi = *(const v8bf*)(p + kbase + 16);   // K = kbase+16 .. kbase+23
  return __builtin_shufflevector(lo, hi, 0,1,2,3,4,5,6,7,8,9,10,11,12,13,14,15);
}
// B: 32x16 (KxN) bf16 tile; lane = N (lane&15), K = kb..kb+7, kb+16..kb+23 with kb = 8*(lane>=16).
__device__ __forceinline__ v16bf load_frag_b(const bf16_t* lds, int col0, int ldn) {
  const int lane = threadIdx.x & 31;
  const int n = lane & 15;
  const int kb = (lane >> 4) << 3;
  v16bf r;
#pragma unroll
  for (int s = 0; s < 8; ++s) {
    r[s]     = lds[(kb + s) * ldn + col0 + n];
    r[8 + s] = lds[(16 + kb + s) * ldn + col0 + n];
  }
  return r;
}
// C/D f32 16x16: lane = N (0..15 twice), VGPR r -> M = r + 8*(lane>=16).
template <int ACT, typename OutT>
__device__ __forceinline__ void store_tile(OutT* C, int ldc, int row0, int col0,
                                           v8f acc) {
  const int lane = threadIdx.x & 31;
  const int n    = lane & 15;
  const int mo   = (lane >> 4) << 3;
#pragma unroll
  for (int r = 0; r < 8; ++r) {
    float v = acc[r];
    if (ACT == 1) v = fmaxf(v, 0.0f);
    if (ACT == 2) v = 1.0f / (1.0f + __expf(-v));
    C[(long)(row0 + mo + r) * ldc + col0 + n] = (OutT)v;
  }
}

// ---- small prep kernels ---------------------------------------------------
__global__ void cvt_f32_bf16_kernel(const float* __restrict__ s,
                                    bf16_t* __restrict__ d, int n) {
  int i = blockIdx.x * 256 + threadIdx.x;
  if (i < n) d[i] = (bf16_t)s[i];
}
__global__ void gather_cvt_rows_kernel(const float* __restrict__ src,
                                       const int* __restrict__ idx,
                                       bf16_t* __restrict__ dst, int ncols) {
  const int r = blockIdx.x, row = idx[r];
  for (int c = threadIdx.x; c < ncols; c += blockDim.x)
    dst[(long)r * ncols + c] = (bf16_t)src[(long)row * ncols + c];
}
__global__ void gather_rows_bf16_kernel(const bf16_t* __restrict__ src,
                                        const int* __restrict__ idx,
                                        bf16_t* __restrict__ dst, int ncols,
                                        int src_ld, int dst_ld) {
  const int r = blockIdx.x, row = idx[r];
  for (int c = threadIdx.x; c < ncols; c += blockDim.x)
    dst[(long)r * dst_ld + c] = src[(long)row * src_ld + c];
}

// ---- GEMM 1: C_bf16 = A_f32 @ B_bf16  (mask aggregation) ------------------
// Grid: (M/64, N/64). 256 threads = 8 waves (2x4); wave tile 32x16.
// Double-buffered LDS; A/B for iter k+1 issued before compute of iter k.
__global__ __launch_bounds__(256) void gemm_a32_kernel(
    const float* __restrict__ A, int lda, const bf16_t* __restrict__ B,
    int ldb, bf16_t* __restrict__ C, int ldc, int K) {
  __shared__ __align__(16) bf16_t sA[2][64 * LDAK];
  __shared__ __align__(16) bf16_t sB[2][32 * LDBN];
  const int t = threadIdx.x, w = t >> 5;
  const int wm = w >> 2, wn = w & 3;       // 2x4 waves -> 64x64 block
  const long blockM = (long)blockIdx.x * 64;
  const long blockN = (long)blockIdx.y * 64;

  v8f acc[2] = {};

  const int ar = t >> 2, ac = (t & 3) << 3;   // A: 4 thr/row, 8 floats each
  const int br = t >> 3, bc = (t & 7) << 3;   // B: 8 thr/row, 8 halves each

  const float*  ag = A + (blockM + ar) * (long)lda + ac;   // + k0
  const bf16_t* bg = B + (long)br * ldb + blockN + bc;     // + k0*ldb

  // ---- prologue: fill buffer 0
  float4 a0 = *(const float4*)(ag);
  float4 a1 = *(const float4*)(ag + 4);
  async_copy16(bg, &sB[0][br * LDBN + bc]);
  {
    union { bf16_t h[8]; uint4 u; } pk;
#pragma unroll
    for (int j = 0; j < 4; ++j) { pk.h[j] = (bf16_t)(&a0.x)[j]; pk.h[4 + j] = (bf16_t)(&a1.x)[j]; }
    *(uint4*)&sA[0][ar * LDAK + ac] = pk.u;
  }
  async_wait0();
  __syncthreads();

  for (int k0 = 0; k0 < K; k0 += 32) {
    const int cur = (k0 >> 5) & 1, nxt = cur ^ 1;
    const bool more = (k0 + 32) < K;
    if (more) {  // issue next tile's traffic before computing this one
      async_copy16(bg + (long)(k0 + 32) * ldb, &sB[nxt][br * LDBN + bc]);
      a0 = *(const float4*)(ag + k0 + 32);
      a1 = *(const float4*)(ag + k0 + 36);
      if (k0 + 64 < K)  // warm L2 for the tile after that
        __builtin_prefetch(ag + k0 + 64, 0, 1);
    }

    v16bf af0 = load_frag_a(sA[cur], wm * 32, LDAK);
    v16bf af1 = load_frag_a(sA[cur], wm * 32 + 16, LDAK);
    v16bf bf  = load_frag_b(sB[cur], wn * 16, LDBN);
    acc[0] = __builtin_amdgcn_wmma_f32_16x16x32_bf16(
        false, af0, false, bf, (short)0, acc[0], false, false);
    acc[1] = __builtin_amdgcn_wmma_f32_16x16x32_bf16(
        false, af1, false, bf, (short)0, acc[1], false, false);

    if (more) {
      union { bf16_t h[8]; uint4 u; } pk;
#pragma unroll
      for (int j = 0; j < 4; ++j) { pk.h[j] = (bf16_t)(&a0.x)[j]; pk.h[4 + j] = (bf16_t)(&a1.x)[j]; }
      *(uint4*)&sA[nxt][ar * LDAK + ac] = pk.u;
      async_wait0();
    }
    __syncthreads();
  }
#pragma unroll
  for (int mi = 0; mi < 2; ++mi)
    store_tile<0, bf16_t>(C, ldc, (int)blockM + wm * 32 + mi * 16,
                          (int)blockN + wn * 16, acc[mi]);
}

// ---- GEMM 2: C = act(A_bf16 @ B_bf16)  (dense layers) ---------------------
// Grid: (M/128, N/64). 256 threads = 8 waves; wave tile 32x32.
template <int ACT, typename OutT>
__global__ __launch_bounds__(256) void gemm_bf16_kernel(
    const bf16_t* __restrict__ A, int lda, const bf16_t* __restrict__ B,
    int ldb, OutT* __restrict__ C, int ldc, int K) {
  __shared__ __align__(16) bf16_t sA[128 * LDAK];
  __shared__ __align__(16) bf16_t sB[32 * LDBN];
  const int t = threadIdx.x, w = t >> 5;
  const int wm = w >> 1, wn = w & 1;       // 4x2 waves -> 128x64 block
  const long blockM = (long)blockIdx.x * 128;
  const long blockN = (long)blockIdx.y * 64;

  v8f acc[2][2] = {};

  const int ar = t >> 2, ac = (t & 3) << 3;   // A: 4 thr/row, 8 halves each
  const int br = t >> 3, bc = (t & 7) << 3;   // B: 8 thr/row, 8 halves each

  for (int k0 = 0; k0 < K; k0 += 32) {
#pragma unroll
    for (int i = 0; i < 2; ++i) {
      const int r = ar + 64 * i;
      async_copy16(A + (blockM + r) * (long)lda + k0 + ac, &sA[r * LDAK + ac]);
    }
    async_copy16(B + (long)(k0 + br) * ldb + blockN + bc, &sB[br * LDBN + bc]);
    async_wait0();
    __syncthreads();

    v16bf af[2], bfr[2];
#pragma unroll
    for (int mi = 0; mi < 2; ++mi)
      af[mi] = load_frag_a(sA, wm * 32 + mi * 16, LDAK);
#pragma unroll
    for (int ni = 0; ni < 2; ++ni)
      bfr[ni] = load_frag_b(sB, wn * 32 + ni * 16, LDBN);
#pragma unroll
    for (int ni = 0; ni < 2; ++ni)
#pragma unroll
      for (int mi = 0; mi < 2; ++mi)
        acc[mi][ni] = __builtin_amdgcn_wmma_f32_16x16x32_bf16(
            false, af[mi], false, bfr[ni], (short)0, acc[mi][ni], false, false);
    __syncthreads();
  }
#pragma unroll
  for (int mi = 0; mi < 2; ++mi)
#pragma unroll
    for (int ni = 0; ni < 2; ++ni)
      store_tile<ACT, OutT>(C, ldc, (int)blockM + wm * 32 + mi * 16,
                            (int)blockN + wn * 32 + ni * 16, acc[mi][ni]);
}

// ---------------------------------------------------------------------------
extern "C" void kernel_launch(void* const* d_in, const int* in_sizes, int n_in,
                              void* d_out, int out_size, void* d_ws,
                              size_t ws_size, hipStream_t stream) {
  const float* in_features = (const float*)d_in[0];  // 200000 x 128
  const int*   idx0        = (const int*)d_in[1];    // 16384
  const int*   cur1        = (const int*)d_in[2];    // 8192
  const int*   cur2        = (const int*)d_in[3];    // 1024
  const float* mask1       = (const float*)d_in[4];  // 8192 x 16384
  const float* mask2       = (const float*)d_in[5];  // 1024 x 8192
  const float* W1          = (const float*)d_in[6];  // 256 x 256
  const float* W2          = (const float*)d_in[7];  // 512 x 256
  const float* W_out       = (const float*)d_in[8];  // 256 x 64
  float* out = (float*)d_out;                        // 1024 x 64

  char* ws = (char*)d_ws;
  bf16_t* f0  = (bf16_t*)(ws);                 // 16384 x 128
  bf16_t* X1  = (bf16_t*)(ws + (4u << 20));    // 8192 x 256  [f0[cur1] | agg1]
  bf16_t* f1  = (bf16_t*)(ws + (8u << 20));    // 8192 x 256
  bf16_t* X2  = (bf16_t*)(ws + (12u << 20));   // 1024 x 512  [f1[cur2] | agg2]
  bf16_t* f2  = (bf16_t*)(ws + (14u << 20));   // 1024 x 256
  bf16_t* W1b = (bf16_t*)(ws + (15u << 20));   // 256 x 256
  bf16_t* W2b = W1b + 256 * 256;               // 512 x 256
  bf16_t* Wob = W2b + 512 * 256;               // 256 x 64

  // weights -> bf16
  cvt_f32_bf16_kernel<<<256, 256, 0, stream>>>(W1, W1b, 256 * 256);
  cvt_f32_bf16_kernel<<<512, 256, 0, stream>>>(W2, W2b, 512 * 256);
  cvt_f32_bf16_kernel<<<64, 256, 0, stream>>>(W_out, Wob, 256 * 64);
  // f0 = bf16(in_features[idx0])
  gather_cvt_rows_kernel<<<16384, 128, 0, stream>>>(in_features, idx0, f0, 128);
  // X1[:, :128] = f0[cur1]
  gather_rows_bf16_kernel<<<8192, 128, 0, stream>>>(f0, cur1, X1, 128, 128, 256);
  // X1[:, 128:] = agg1 = mask1 @ f0        (M=8192, K=16384, N=128)
  gemm_a32_kernel<<<dim3(128, 2), 256, 0, stream>>>(mask1, 16384, f0, 128,
                                                    X1 + 128, 256, 16384);
  // f1 = relu(X1 @ W1)                     (M=8192, K=256, N=256)
  gemm_bf16_kernel<1, bf16_t><<<dim3(64, 4), 256, 0, stream>>>(X1, 256, W1b,
                                                               256, f1, 256, 256);
  // X2[:, :256] = f1[cur2]
  gather_rows_bf16_kernel<<<1024, 256, 0, stream>>>(f1, cur2, X2, 256, 256, 512);
  // X2[:, 256:] = agg2 = mask2 @ f1        (M=1024, K=8192, N=256)
  gemm_a32_kernel<<<dim3(16, 4), 256, 0, stream>>>(mask2, 8192, f1, 256,
                                                   X2 + 256, 512, 8192);
  // f2 = relu(X2 @ W2)                     (M=1024, K=512, N=256)
  gemm_bf16_kernel<1, bf16_t><<<dim3(8, 4), 256, 0, stream>>>(X2, 512, W2b,
                                                              256, f2, 256, 512);
  // out = sigmoid(f2 @ W_out)              (M=1024, K=256, N=64) -> f32
  gemm_bf16_kernel<2, float><<<dim3(8, 1), 256, 0, stream>>>(f2, 256, Wob, 64,
                                                             out, 64, 256);
}
